// MatchingNet_62758062129318
// MI455X (gfx1250) — compile-verified, matrix operand order checked by reading, not run on hardware
//
#include <hip/hip_runtime.h>
#include <hip/hip_bf16.h>

typedef __attribute__((ext_vector_type(16))) __bf16 v16bf;
typedef __attribute__((ext_vector_type(8)))  float  v8f;

union FragBF {
    v16bf v;
    unsigned short u[16];
    uint4 q[2];
};

__device__ __forceinline__ unsigned short f2bf(float f) {
    union { float f; unsigned int u; } x;
    x.f = f;
    unsigned int u = x.u;
    u += 0x7FFFu + ((u >> 16) & 1u);   // round-to-nearest-even
    return (unsigned short)(u >> 16);
}

// ---------------------------------------------------------------------------
// W [IN, E] fp32 (row-major)  ->  Wt [E, IN] bf16 (row-major), LDS transpose
// ---------------------------------------------------------------------------
__global__ void __launch_bounds__(1024) wt_transpose_bf16(
    const float* __restrict__ W, unsigned short* __restrict__ Wt, int IN, int E) {
    __shared__ float tile[32][33];
    const int e0 = blockIdx.x * 32;
    const int k0 = blockIdx.y * 32;
    const int tx = threadIdx.x, ty = threadIdx.y;
    tile[ty][tx] = W[(size_t)(k0 + ty) * E + (e0 + tx)];
    __syncthreads();
    Wt[(size_t)(e0 + ty) * IN + (k0 + tx)] = f2bf(tile[tx][ty]);
}

// ---------------------------------------------------------------------------
// emb = l2norm(X @ Wt^T + b), output bf16 [N, E]
// block = 256 threads (8 waves); wave computes a 16 x 128 strip, K-loop of 32,
// 8 accumulators sharing one A-fragment (8x A reuse).
// ---------------------------------------------------------------------------
__global__ void __launch_bounds__(256) encoder_kernel(
    const float* __restrict__ X, const unsigned short* __restrict__ Wt,
    const float* __restrict__ bias, unsigned short* __restrict__ out,
    int IN, int E) {
    const int lane = threadIdx.x & 31;
    const int wave = threadIdx.x >> 5;
    const int half = lane >> 4;     // 0: lanes 0-15, 1: lanes 16-31
    const int l16  = lane & 15;
    const int m0   = blockIdx.x * 16;

    v8f acc[8];
#pragma unroll
    for (int j = 0; j < 8; ++j)
        acc[j] = (v8f){0.f, 0.f, 0.f, 0.f, 0.f, 0.f, 0.f, 0.f};

    const float* xrow = X + (size_t)(m0 + l16) * IN;

    for (int k0 = 0; k0 < IN; k0 += 32) {
        FragBF a;
        const float* pa = xrow + k0 + half * 8;
        float4 f0 = *(const float4*)(pa);
        float4 f1 = *(const float4*)(pa + 4);
        float4 f2 = *(const float4*)(pa + 16);
        float4 f3 = *(const float4*)(pa + 20);
        a.u[0]  = f2bf(f0.x); a.u[1]  = f2bf(f0.y); a.u[2]  = f2bf(f0.z); a.u[3]  = f2bf(f0.w);
        a.u[4]  = f2bf(f1.x); a.u[5]  = f2bf(f1.y); a.u[6]  = f2bf(f1.z); a.u[7]  = f2bf(f1.w);
        a.u[8]  = f2bf(f2.x); a.u[9]  = f2bf(f2.y); a.u[10] = f2bf(f2.z); a.u[11] = f2bf(f2.w);
        a.u[12] = f2bf(f3.x); a.u[13] = f2bf(f3.y); a.u[14] = f2bf(f3.z); a.u[15] = f2bf(f3.w);

#pragma unroll
        for (int j = 0; j < 8; ++j) {
            const int n0 = (wave + 8 * j) * 16;
            FragBF b;
            const unsigned short* pb = Wt + (size_t)(n0 + l16) * IN + k0 + half * 8;
            b.q[0] = *(const uint4*)(pb);
            b.q[1] = *(const uint4*)(pb + 16);
            acc[j] = __builtin_amdgcn_wmma_f32_16x16x32_bf16(
                false, a.v, false, b.v, (short)0, acc[j], false, false);
        }
    }

    // bias add (bias depends only on column N = n0 + l16)
#pragma unroll
    for (int j = 0; j < 8; ++j) {
        const int n0 = (wave + 8 * j) * 16;
        const float bb = bias[n0 + l16];
#pragma unroll
        for (int r = 0; r < 8; ++r) acc[j][r] += bb;
    }

    // per-row sum of squares: lane l, reg r holds element (M = r + 8*half, N = l16)
    __shared__ float norm[16];
    if (threadIdx.x < 16) norm[threadIdx.x] = 0.0f;
    __syncthreads();
#pragma unroll
    for (int r = 0; r < 8; ++r) {
        float sq = 0.f;
#pragma unroll
        for (int j = 0; j < 8; ++j) sq += acc[j][r] * acc[j][r];
        for (int off = 1; off < 16; off <<= 1) sq += __shfl_xor(sq, off, 32);
        if (l16 == 0) atomicAdd(&norm[r + 8 * half], sq);
    }
    __syncthreads();

    float inv[8];
#pragma unroll
    for (int r = 0; r < 8; ++r) {
        const float nn = sqrtf(norm[r + 8 * half]);
        inv[r] = 1.0f / fmaxf(nn, 1e-12f);
    }
#pragma unroll
    for (int j = 0; j < 8; ++j) {
        const int n0 = (wave + 8 * j) * 16;
#pragma unroll
        for (int r = 0; r < 8; ++r) {
            const int row = m0 + r + 8 * half;
            out[(size_t)row * E + n0 + l16] = f2bf(acc[j][r] * inv[r]);
        }
    }
}

// ---------------------------------------------------------------------------
// Fused streaming attention:
//   sims = q_emb @ s_emb^T (bf16 WMMA, fp32 acc); since |sim| <= 1, softmax
//   needs no max-trick: preds_unnorm[q][label[s]] += exp(sim), denominator is
//   the class-sum. Single streaming pass, no sims matrix anywhere.
//
// block = 256 threads (8 waves), 32 query rows (2 q-tiles).
// Support streamed in 128-row chunks staged in LDS via CDNA5 async loads
// (global_load_async_to_lds_b128, ASYNCcnt). Rows padded by 8 ushorts so the
// 16 lanes of a ds_load_b128 B-fragment stride 4 banks -> conflict-free.
// Each wave: one q-tile x 2 support tiles register-blocked on one A-frag.
// ---------------------------------------------------------------------------
#define ATT_BQ 32     // query rows per block
#define ATT_SK 128    // support rows staged per chunk
#define ATT_PAD 8     // ushort padding per staged row

__global__ void __launch_bounds__(256) attention_kernel(
    const unsigned short* __restrict__ q_emb, const unsigned short* __restrict__ s_emb,
    const int* __restrict__ labels, float* __restrict__ out,
    int NS, int E, int C) {
    extern __shared__ unsigned char smem_raw[];
    const int EP = E + ATT_PAD;                                    // padded row (ushorts)
    unsigned short* sK = (unsigned short*)smem_raw;                // [ATT_SK][EP]
    float* cls = (float*)(smem_raw + (size_t)ATT_SK * EP * 2);     // [ATT_BQ][C]

    const int tid  = threadIdx.x;
    const int lane = tid & 31;
    const int wave = tid >> 5;
    const int half = lane >> 4;
    const int l16  = lane & 15;
    const int q0   = blockIdx.x * ATT_BQ;
    const int qti  = wave >> 2;          // q-tile 0..1
    const int tp   = wave & 3;           // support tile pair {tp, tp+4} in chunk

    for (int i = tid; i < ATT_BQ * C; i += blockDim.x) cls[i] = 0.0f;
    __syncthreads();

    const unsigned short* qrow = q_emb + (size_t)(q0 + qti * 16 + l16) * E + half * 8;
    const unsigned sK_base = (unsigned)(uintptr_t)sK;
    const int vecs_per_row = E / 8;      // 16B units per row

    for (int c0 = 0; c0 < NS; c0 += ATT_SK) {
        // ---- stage chunk into LDS via async DMA (row-padded) ----
        {
            const unsigned short* src = s_emb + (size_t)c0 * E;
            for (int i = tid; i < ATT_SK * vecs_per_row; i += blockDim.x) {
                const int row = i / vecs_per_row;
                const int col = i - row * vecs_per_row;
                const unsigned lds_addr = sK_base + (unsigned)(row * EP + col * 8) * 2u;
                const unsigned short* g = src + (size_t)row * E + col * 8;
                asm volatile("global_load_async_to_lds_b128 %0, %1, off"
                             :: "v"(lds_addr), "v"(g) : "memory");
            }
            asm volatile("s_wait_asynccnt 0x0" ::: "memory");
        }
        __syncthreads();

        // ---- 2 support tiles per wave, sharing each A-fragment ----
        v8f acc[2];
        acc[0] = (v8f){0.f, 0.f, 0.f, 0.f, 0.f, 0.f, 0.f, 0.f};
        acc[1] = (v8f){0.f, 0.f, 0.f, 0.f, 0.f, 0.f, 0.f, 0.f};
        const unsigned short* b0 = sK + (size_t)(tp * 16 + l16) * EP + half * 8;
        const unsigned short* b1 = sK + (size_t)((tp + 4) * 16 + l16) * EP + half * 8;

        for (int k0 = 0; k0 < E; k0 += 32) {
            FragBF a, bA, bB;
            a.q[0]  = *(const uint4*)(qrow + k0);
            a.q[1]  = *(const uint4*)(qrow + k0 + 16);
            bA.q[0] = *(const uint4*)(b0 + k0);
            bA.q[1] = *(const uint4*)(b0 + k0 + 16);
            bB.q[0] = *(const uint4*)(b1 + k0);
            bB.q[1] = *(const uint4*)(b1 + k0 + 16);
            acc[0] = __builtin_amdgcn_wmma_f32_16x16x32_bf16(
                false, a.v, false, bA.v, (short)0, acc[0], false, false);
            acc[1] = __builtin_amdgcn_wmma_f32_16x16x32_bf16(
                false, a.v, false, bB.v, (short)0, acc[1], false, false);
        }

        // ---- exp + label-bucketed accumulation (|sim|<=1 -> no max-trick) ----
#pragma unroll
        for (int u = 0; u < 2; ++u) {
            const int scol = c0 + (tp + 4 * u) * 16 + l16;
            const unsigned lab = (unsigned)labels[scol] % (unsigned)C;
            float* clsrow = cls + (size_t)(qti * 16) * C + lab;
#pragma unroll
            for (int r = 0; r < 8; ++r) {
                const float w = __expf(acc[u][r]);
                const int m = r + 8 * half;
                atomicAdd(clsrow + m * C, w);
            }
        }
        __syncthreads();   // protect sK before next chunk's staging
    }

    // ---- normalize: denominator is the class-sum of each row ----
    for (int i = tid; i < ATT_BQ; i += blockDim.x) {
        float denom = 0.f;
        for (int c = 0; c < C; ++c) denom += cls[i * C + c];
        const float inv = 1.0f / denom;
        float* orow = out + (size_t)(q0 + i) * C;
        for (int c = 0; c < C; ++c) orow[c] = cls[i * C + c] * inv;
    }
}

// ---------------------------------------------------------------------------
extern "C" void kernel_launch(void* const* d_in, const int* in_sizes, int n_in,
                              void* d_out, int out_size, void* d_ws, size_t ws_size,
                              hipStream_t stream) {
    const float* support = (const float*)d_in[0];
    const float* query   = (const float*)d_in[1];
    const float* W       = (const float*)d_in[2];
    const float* b       = (const float*)d_in[3];
    const int*   labels  = (const int*)d_in[4];

    const int E  = in_sizes[3];            // 1024
    const int IN = in_sizes[2] / E;        // 2048
    const int NS = in_sizes[0] / IN;       // 4096
    const int NQ = in_sizes[1] / IN;       // 8192
    const int C  = out_size / NQ;          // 64

    unsigned short* Wt    = (unsigned short*)d_ws;            // [E, IN] bf16, 4 MB
    unsigned short* s_emb = Wt + (size_t)E * IN;              // [NS, E] bf16, 8 MB
    unsigned short* q_emb = s_emb + (size_t)NS * E;           // [NQ, E] bf16, 16 MB

    wt_transpose_bf16<<<dim3(E / 32, IN / 32), dim3(32, 32), 0, stream>>>(W, Wt, IN, E);

    encoder_kernel<<<NS / 16, 256, 0, stream>>>(support, Wt, b, s_emb, IN, E);
    encoder_kernel<<<NQ / 16, 256, 0, stream>>>(query, Wt, b, q_emb, IN, E);

    const size_t smem = (size_t)ATT_SK * (E + ATT_PAD) * 2 +
                        (size_t)ATT_BQ * C * sizeof(float);   // ~272 KB of 320 KB LDS
    attention_kernel<<<NQ / ATT_BQ, 256, smem, stream>>>(q_emb, s_emb, labels,
                                                         (float*)d_out, NS, E, C);
}
